// SA2_82875688943934
// MI455X (gfx1250) — matrix operand order repaired and checked by reading, not compiled, not run on hardware
//
#include <hip/hip_runtime.h>
#include <hip/hip_bf16.h>
#include <cstddef>

typedef __attribute__((ext_vector_type(16))) __bf16 v16bf;
typedef __attribute__((ext_vector_type(8)))  float  v8f;

#define Bn   4
#define Cc   256
#define Dqk  32
#define Nn   4096   // H*W

// ---------------------------------------------------------------------------
// Kernel 1: 1x1-conv projections.  q,k stored transposed [N,32] bf16 (WMMA
// A/B fragment friendly), v stored [C,N] bf16.
// grid = B*(N/64) blocks of 256 threads, 64 KB dynamic LDS.
// ---------------------------------------------------------------------------
__global__ void sa2_proj(const float* __restrict__ saE,
                         const float* __restrict__ saD,
                         const float* __restrict__ Wq, const float* __restrict__ bq,
                         const float* __restrict__ Wk, const float* __restrict__ bk,
                         const float* __restrict__ Wv, const float* __restrict__ bv,
                         __bf16* __restrict__ qT, __bf16* __restrict__ kT,
                         __bf16* __restrict__ vB) {
  extern __shared__ float tile[];            // [256][64] f32 = 64 KB
  const int b  = blockIdx.x >> 6;
  const int n0 = (blockIdx.x & 63) * 64;
  const int tid = threadIdx.x;
  const int px = tid & 63;
  const int og = tid >> 6;                   // 0..3, uniform per wave32

  // ---- phase 1: q,k from sa_D ----
  for (int i = tid; i < Cc * 64; i += 256) {
    int c = i >> 6, p = i & 63;
    tile[i] = saD[(size_t)b * Cc * Nn + (size_t)c * Nn + n0 + p];
  }
  __syncthreads();

  for (int o = og * 8; o < og * 8 + 8; ++o) {
    float aq = bq[o], ak = bk[o];
    for (int c = 0; c < Cc; ++c) {
      float d = tile[c * 64 + px];
      aq = fmaf(Wq[o * Cc + c], d, aq);
      ak = fmaf(Wk[o * Cc + c], d, ak);
    }
    size_t row = ((size_t)b * Nn + n0 + px) * Dqk;
    qT[row + o] = (__bf16)aq;
    kT[row + o] = (__bf16)ak;
  }
  __syncthreads();

  // ---- phase 2: v from sa_E (reuse LDS) ----
  for (int i = tid; i < Cc * 64; i += 256) {
    int c = i >> 6, p = i & 63;
    tile[i] = saE[(size_t)b * Cc * Nn + (size_t)c * Nn + n0 + p];
  }
  __syncthreads();

  for (int o = og * 64; o < og * 64 + 64; ++o) {
    float av = bv[o];
    for (int c = 0; c < Cc; ++c)
      av = fmaf(Wv[o * Cc + c], tile[c * 64 + px], av);
    vB[((size_t)b * Cc + o) * Nn + n0 + px] = (__bf16)av;
  }
}

// ---------------------------------------------------------------------------
// Kernel 2: energy = q^T k (WMMA, K=32 in one issue) + row softmax.
// One block per (b, 16-row tile). 8 waves x 512 columns each -> 16x4096 f32
// row buffer in LDS (256 KB). Writes attention f32 (output) + bf16 (ws).
// ---------------------------------------------------------------------------
__global__ void sa2_energy_softmax(const __bf16* __restrict__ qT,
                                   const __bf16* __restrict__ kT,
                                   float* __restrict__ attF,
                                   __bf16* __restrict__ attB) {
  extern __shared__ float elds[];            // [16][4096] f32
  const int b  = blockIdx.x >> 8;            // N/16 = 256 tiles per batch
  const int n0 = (blockIdx.x & 255) * 16;
  const int tid  = threadIdx.x;
  const int w    = tid >> 5;
  const int lane = tid & 31;
  const int lr   = lane & 15;
  const int hi   = lane >> 4;

  // A fragment: 16 rows of q^T (row n, K = d 0..31)
  v16bf a;
  {
    const __bf16* qrow = qT + ((size_t)b * Nn + n0 + lr) * Dqk;
#pragma unroll
    for (int j = 0; j < 16; ++j) {
      int kk = j + hi * 16;
      a[j] = qrow[kk];
    }
  }

  const v8f czero = {};
  for (int i = 0; i < 32; ++i) {
    int m0 = w * 512 + i * 16;
    // B fragment: 16 columns of k (col m, K = d), contiguous per lane
    const __bf16* krow = kT + ((size_t)b * Nn + m0 + lr) * Dqk + hi * 16;
    v16bf bb;
#pragma unroll
    for (int j = 0; j < 16; ++j) bb[j] = krow[j];

    v8f d = __builtin_amdgcn_wmma_f32_16x16x32_bf16(
        false, a, false, bb, (short)0, czero, false, false);

    int rb = hi * 8;
#pragma unroll
    for (int r = 0; r < 8; ++r)
      elds[(rb + r) * Nn + m0 + lr] = d[r];
  }
  __syncthreads();

  // softmax: wave w handles rows 2w, 2w+1
  for (int rr = 0; rr < 2; ++rr) {
    int r = w * 2 + rr;
    float mx = -3.0e38f;
    for (int j = lane; j < Nn; j += 32) mx = fmaxf(mx, elds[r * Nn + j]);
#pragma unroll
    for (int off = 16; off >= 1; off >>= 1) mx = fmaxf(mx, __shfl_xor(mx, off, 32));
    float s = 0.f;
    for (int j = lane; j < Nn; j += 32) {
      float e = __expf(elds[r * Nn + j] - mx);
      elds[r * Nn + j] = e;
      s += e;
    }
#pragma unroll
    for (int off = 16; off >= 1; off >>= 1) s += __shfl_xor(s, off, 32);
    float inv = 1.0f / s;
    size_t rowbase = ((size_t)b * Nn + n0 + r) * (size_t)Nn;
    for (int j = lane; j < Nn; j += 32) {
      float v = elds[r * Nn + j] * inv;
      attF[rowbase + j] = v;
      attB[rowbase + j] = (__bf16)v;
    }
  }
}

// ---------------------------------------------------------------------------
// Kernel 3: out = gamma * (V @ att^T) + sa_E.  bf16 WMMA GEMM, K=4096.
// Block tile 128c x 64n, 8 waves (one 16c strip each, 4 WMMA tiles wide).
// The 64n x 32m attention tile is staged in LDS once per block via the CDNA5
// async global->LDS path (ASYNCcnt), double-buffered; padded row stride of
// 40 bf16 (20 dwords) keeps the b128 fragment reads bank-conflict free.
// grid = (N/64, C/128, B), 256 threads.
// ---------------------------------------------------------------------------
#define LDS_STRIDE 40                        // bf16 elements per staged row

__global__ void sa2_out_gemm(const __bf16* __restrict__ vB,
                             const __bf16* __restrict__ attB,
                             const float* __restrict__ saE,
                             const float* __restrict__ gamma,
                             float* __restrict__ outF) {
  __shared__ __bf16 attS[2][64 * LDS_STRIDE];   // 2 x 5 KB

  const int n0 = blockIdx.x * 64;
  const int c0 = blockIdx.y * 128;
  const int b  = blockIdx.z;
  const int tid  = threadIdx.x;
  const int w    = tid >> 5;                 // 0..7 -> c strip
  const int lane = tid & 31;
  const int lr   = lane & 15;
  const int hi   = lane >> 4;

  // staging assignment: each thread async-copies one 16B chunk per K step
  const int srow = tid >> 2;                 // 0..63  (n within tile)
  const int sseg = tid & 3;                  // 0..3   (8-elem chunk of m)
  const __bf16* gsrcBase =
      attB + ((size_t)b * Nn + n0 + srow) * (size_t)Nn + sseg * 8;
  const unsigned ldsOff0 =
      (unsigned)(unsigned long long)&attS[0][srow * LDS_STRIDE + sseg * 8];
  const unsigned ldsOff1 =
      (unsigned)(unsigned long long)&attS[1][srow * LDS_STRIDE + sseg * 8];

  const int crow = c0 + w * 16 + lr;         // A-fragment row (c)
  const __bf16* vrow = vB + ((size_t)b * Cc + crow) * Nn;

  // prologue: stage first tile into buffer 0
  asm volatile("global_load_async_to_lds_b128 %0, %1, off"
               :: "v"(ldsOff0), "v"((unsigned long long)gsrcBase)
               : "memory");

  v8f acc[4] = {{}, {}, {}, {}};
  for (int it = 0; it < Nn / 32; ++it) {
    const int m0 = it * 32;
    if (it + 1 < Nn / 32) {               // stage next tile into other buffer
      unsigned ldsOff = ((it + 1) & 1) ? ldsOff1 : ldsOff0;
      asm volatile("global_load_async_to_lds_b128 %0, %1, off"
                   :: "v"(ldsOff), "v"((unsigned long long)(gsrcBase + m0 + 32))
                   : "memory");
      asm volatile("s_wait_asynccnt 0x1" ::: "memory");
    } else {
      asm volatile("s_wait_asynccnt 0x0" ::: "memory");
    }
    __syncthreads();                      // current buffer fully staged

    __builtin_prefetch(vrow + m0 + 128, 0, 1);   // global_prefetch A stream

    v16bf a;
#pragma unroll
    for (int j = 0; j < 16; ++j) {
      int kk = j + hi * 16;
      a[j] = vrow[m0 + kk];
    }

    const __bf16* as = attS[it & 1];
#pragma unroll
    for (int t = 0; t < 4; ++t) {
      const __bf16* bsrc = as + (t * 16 + lr) * LDS_STRIDE + hi * 16;
      v16bf bb;
#pragma unroll
      for (int j = 0; j < 16; ++j) bb[j] = bsrc[j];
      acc[t] = __builtin_amdgcn_wmma_f32_16x16x32_bf16(
          false, a, false, bb, (short)0, acc[t], false, false);
    }
    __syncthreads();                      // all reads done before next overwrite
  }

  const float g = gamma[0];
  const int rb = hi * 8;
#pragma unroll
  for (int t = 0; t < 4; ++t) {
#pragma unroll
    for (int r = 0; r < 8; ++r) {
      int cc = c0 + w * 16 + rb + r;
      int nn = n0 + t * 16 + lr;
      size_t idx = ((size_t)b * Cc + cc) * Nn + nn;
      outF[idx] = fmaf(g, acc[t][r], saE[idx]);
    }
  }
}

// ---------------------------------------------------------------------------
extern "C" void kernel_launch(void* const* d_in, const int* in_sizes, int n_in,
                              void* d_out, int out_size, void* d_ws, size_t ws_size,
                              hipStream_t stream) {
  const float* saE   = (const float*)d_in[0];
  const float* saD   = (const float*)d_in[1];
  const float* Wq    = (const float*)d_in[2];
  const float* bq    = (const float*)d_in[3];
  const float* Wk    = (const float*)d_in[4];
  const float* bk    = (const float*)d_in[5];
  const float* Wv    = (const float*)d_in[6];
  const float* bv    = (const float*)d_in[7];
  const float* gamma = (const float*)d_in[8];

  // workspace layout (bytes): qT(1MB) | kT(1MB) | vB(8MB) | attB(128MB)
  char* ws = (char*)d_ws;
  __bf16* qT   = (__bf16*)(ws);
  __bf16* kT   = (__bf16*)(ws + (size_t)Bn * Nn * Dqk * 2);
  __bf16* vB   = (__bf16*)(ws + (size_t)2 * Bn * Nn * Dqk * 2);
  __bf16* attB = (__bf16*)(ws + (size_t)2 * Bn * Nn * Dqk * 2 + (size_t)Bn * Cc * Nn * 2);

  float* outF = (float*)d_out;                          // [B,C,H,W]
  float* attF = outF + (size_t)Bn * Cc * Nn;            // [B,N,N]

  // allow large dynamic LDS (WGP has 320 KB)
  static bool attrDone = false;  // host-side only; does not affect device work
  if (!attrDone) {
    hipFuncSetAttribute((const void*)sa2_energy_softmax,
                        hipFuncAttributeMaxDynamicSharedMemorySize, 16 * Nn * 4);
    hipFuncSetAttribute((const void*)sa2_proj,
                        hipFuncAttributeMaxDynamicSharedMemorySize, Cc * 64 * 4);
    attrDone = true;
  }

  sa2_proj<<<dim3(Bn * (Nn / 64)), 256, Cc * 64 * 4, stream>>>(
      saE, saD, Wq, bq, Wk, bk, Wv, bv, qT, kT, vB);

  sa2_energy_softmax<<<dim3(Bn * (Nn / 16)), 256, 16 * Nn * 4, stream>>>(
      qT, kT, attF, attB);

  sa2_out_gemm<<<dim3(Nn / 64, Cc / 128, Bn), 256, 0, stream>>>(
      vB, attB, saE, gamma, outF);
}
